// dyAbModel_14233521619126
// MI455X (gfx1250) — matrix operand order repaired and checked by reading, not compiled, not run on hardware
//
#include <hip/hip_runtime.h>
#include <hip/hip_bf16.h>
#include <cstdint>

typedef __bf16 bf16;
typedef __attribute__((ext_vector_type(16))) __bf16 v16bf;
typedef __attribute__((ext_vector_type(8)))  __bf16 v8bf;
typedef __attribute__((ext_vector_type(8)))  float  v8f;

#define NB    32
#define NN    1024
#define NCH   14
#define NKNN  9
#define NE    128
#define NHID  128
#define NCLS  25
#define BNR   (NB*NN)         // 32768 node rows

__device__ __forceinline__ float siluf(float x){ return x * (1.0f/(1.0f+__expf(-x))); }
__device__ __forceinline__ float sigmf(float x){ return 1.0f/(1.0f+__expf(-x)); }

// ---- CDNA5 async global->LDS staging (ASYNCcnt path) -----------------------
// Each active lane issues a 16B async copy; drained with s_wait_asynccnt 0.
__device__ __forceinline__ void stage_w_async(bf16* lds, const bf16* g, int bytes,
                                              int tid, int nth){
  uint32_t lbase = (uint32_t)(uintptr_t)lds;          // LDS aperture: addr[31:0] = LDS offset
  const char* gp = (const char*)g;
  for (int o = tid*16; o < bytes; o += nth*16){
    asm volatile("global_load_async_to_lds_b128 %0, %1, off"
                 :: "v"(lbase + (uint32_t)o), "v"(gp + o)
                 : "memory");
  }
}
__device__ __forceinline__ void wait_async_all(){
  asm volatile("s_wait_asynccnt 0x0" ::: "memory");
}

// ---- WMMA fragment loaders (bf16, 16x16x32) -------------------------------
// A 16x32 layout (ISA 7.12.2): lane half h -> K groups [h*8, h*8+8) and [16+h*8, 16+h*8+8)
__device__ __forceinline__ v16bf load_fragA(const bf16* base, int stride, int r, int kbase, int h){
  const bf16* p = base + (size_t)r*stride + kbase + h*8;
  v8bf lo = *(const v8bf*)(p);
  v8bf hi = *(const v8bf*)(p+16);
  v16bf f;
#pragma unroll
  for (int i=0;i<8;++i){ f[i]=lo[i]; f[i+8]=hi[i]; }
  return f;
}
// B 32x16 layout (matches A): lane half h -> K groups [h*8, h*8+8) and [16+h*8, 16+h*8+8), col=lane%16.
// Weights pre-packed transposed (row n, K contiguous) -> two 16B reads.
__device__ __forceinline__ v16bf load_fragB(const bf16* wt, int stride, int c, int kbase, int h){
  const bf16* p = wt + (size_t)c*stride + kbase + h*8;
  v8bf lo = *(const v8bf*)(p);
  v8bf hi = *(const v8bf*)(p+16);
  v16bf f;
#pragma unroll
  for (int i=0;i<8;++i){ f[i]=lo[i]; f[i+8]=hi[i]; }
  return f;
}

template<int CT>
__device__ __forceinline__ void zeroacc(v8f* a){
#pragma unroll
  for (int i=0;i<CT;++i)
#pragma unroll
    for (int j=0;j<8;++j) a[i][j]=0.0f;
}

// 16-row x CT*16-col tile GEMM, K = KS*32, A row-major (bf16), WT packed transposed.
template<int KS, int CT>
__device__ __forceinline__ void wmma_rowtile(const bf16* A, int sA, const bf16* WT, int sW,
                                             v8f* acc, int lane){
  const int rc = lane & 15, h = lane >> 4;
#pragma unroll
  for (int ks=0; ks<KS; ++ks){
    v16bf a = load_fragA(A, sA, rc, ks*32, h);
#pragma unroll
    for (int ct=0; ct<CT; ++ct){
      v16bf b = load_fragB(WT, sW, ct*16+rc, ks*32, h);
      acc[ct] = __builtin_amdgcn_wmma_f32_16x16x32_bf16(false, a, false, b,
                                                        (short)0, acc[ct], false, false);
    }
  }
}

// ---- weight pre-pack: out[n*kpad + k] = W[(k0+k)*ldW + n0+n] (bf16, zero-padded)
__global__ void pack_kernel(const float* __restrict__ W, int ldW, int k0, int krows,
                            int n0, int nvalid, bf16* __restrict__ out, int nout, int kpad){
  int t = blockIdx.x*256 + threadIdx.x;
  if (t >= nout*kpad) return;
  int n_ = t / kpad, k_ = t % kpad;
  float v = (n_ < nvalid && k_ < krows) ? W[(size_t)(k0 + k_)*ldW + (n0 + n_)] : 0.0f;
  out[t] = (bf16)v;
}

// ---- KNN over ca = X[:,:,1,:] ----------------------------------------------
__global__ void knn_kernel(const float* __restrict__ X, int* __restrict__ idx){
  __shared__ float sca[NN*3];
  const int b = blockIdx.y, tid = threadIdx.x;
  for (int j = tid; j < NN; j += 256){
    const float* p = X + ((size_t)(b*NN + j)*NCH + 1)*3;
    sca[j*3+0]=p[0]; sca[j*3+1]=p[1]; sca[j*3+2]=p[2];
  }
  __syncthreads();
  const int n = blockIdx.x*256 + tid;
  const float q0=sca[n*3], q1=sca[n*3+1], q2=sca[n*3+2];
  float bd[NKNN]; int bi[NKNN];
#pragma unroll
  for (int k=0;k<NKNN;++k){ bd[k]=3.0e38f; bi[k]=0; }
  for (int j=0;j<NN;++j){
    if (j==n) continue;                       // diag + 1e10 => self excluded
    float dx=q0-sca[j*3], dy=q1-sca[j*3+1], dz=q2-sca[j*3+2];
    float d2 = dx*dx+dy*dy+dz*dz;
    if (d2 < bd[NKNN-1]){
      int p = NKNN-1;
      while (p>0 && bd[p-1] > d2){ bd[p]=bd[p-1]; bi[p]=bi[p-1]; --p; }
      bd[p]=d2; bi[p]=j;
    }
  }
#pragma unroll
  for (int k=0;k<NKNN;++k) idx[((size_t)(b*NN)+n)*NKNN + k] = bi[k];
}

// ---- Hh = emb[S] + time_emb(t); also silu(Hh) bf16 --------------------------
// smem: sbuf(4*16*128 bf16) | w1(128*128) | w2(128*128)  = 81920 B
#define HH_SMEM ((size_t)(4*16*128 + 2*128*128)*2)
__global__ void hh_kernel(const float* __restrict__ t, const int* __restrict__ S,
                          const float* __restrict__ emb,
                          const float* __restrict__ bt1, const float* __restrict__ bt2,
                          const bf16* __restrict__ WT_t1, const bf16* __restrict__ WT_t2,
                          float* __restrict__ Hh, bf16* __restrict__ sHh){
  extern __shared__ char smem[];
  bf16* sbuf = (bf16*)smem;                 // [4][16][128]
  bf16* w1   = sbuf + 4*16*128;
  bf16* w2   = w1 + 128*128;
  const int tid = threadIdx.x;
  const int wave = tid >> 5, lane = tid & 31;
  const int rc = lane & 15, h = lane >> 4;
  const int rowbase = blockIdx.x*64 + wave*16;
  bf16* myb = sbuf + wave*16*128;

  stage_w_async(w1, WT_t1, 128*128*2, tid, 128);
  stage_w_async(w2, WT_t2, 128*128*2, tid, 128);
  {
    const float tv = t[rowbase + rc];
    const float inv = 1.0f/127.0f;
    const float coeff = -0.5f * 127.0f * 127.0f;
    const int j0 = h*64;
#pragma unroll 8
    for (int j=0;j<64;++j){
      float d = tv - (float)(j0+j)*inv + 1e-6f;
      myb[rc*128 + j0+j] = (bf16)__expf(coeff*d*d);
    }
  }
  wait_async_all();
  __syncthreads();
  v8f acc[8]; zeroacc<8>(acc);
  wmma_rowtile<4,8>(myb, 128, w1, 128, acc, lane);
  __syncthreads();
#pragma unroll
  for (int ct=0; ct<8; ++ct)
#pragma unroll
    for (int r=0;r<8;++r){
      int col = ct*16+rc;
      myb[(r+8*h)*128 + col] = (bf16)fmaxf(acc[ct][r] + bt1[col], 0.0f);   // relu
    }
  __syncthreads();
  zeroacc<8>(acc);
  wmma_rowtile<4,8>(myb, 128, w2, 128, acc, lane);
#pragma unroll
  for (int ct=0; ct<8; ++ct)
#pragma unroll
    for (int r=0;r<8;++r){
      int row = rowbase + r + 8*h, col = ct*16+rc;
      float v = acc[ct][r] + bt2[col] + emb[(size_t)S[row]*NE + col];
      Hh[(size_t)row*NE+col] = v;
      sHh[(size_t)row*NE+col] = (bf16)siluf(v);
    }
}

// ---- A1 = silu(Hh)@We1[:128]+be1 ; A2 = silu(Hh)@We1[128:256] ---------------
#define A1A2_SMEM ((size_t)(2*128*128)*2)
__global__ void a1a2_kernel(const bf16* __restrict__ sHh,
                            const bf16* __restrict__ WT_hi, const bf16* __restrict__ WT_hnb,
                            const float* __restrict__ be1,
                            float* __restrict__ A1p, float* __restrict__ A2){
  extern __shared__ char smem[];
  bf16* w1 = (bf16*)smem;
  bf16* w2 = w1 + 128*128;
  const int tid = threadIdx.x;
  const int wave = tid >> 5, lane = tid & 31;
  const int rc = lane & 15, h = lane >> 4;
  const int rowbase = blockIdx.x*64 + wave*16;
  const bf16* A = sHh + (size_t)rowbase*NE;

  stage_w_async(w1, WT_hi,  128*128*2, tid, 128);
  stage_w_async(w2, WT_hnb, 128*128*2, tid, 128);
  wait_async_all();
  __syncthreads();

  v8f acc[8]; zeroacc<8>(acc);
  wmma_rowtile<4,8>(A, 128, w1, 128, acc, lane);
#pragma unroll
  for (int ct=0; ct<8; ++ct)
#pragma unroll
    for (int r=0;r<8;++r){
      int row = rowbase + r + 8*h, col = ct*16+rc;
      A1p[(size_t)row*NHID+col] = acc[ct][r] + be1[col];
    }
  zeroacc<8>(acc);
  wmma_rowtile<4,8>(A, 128, w2, 128, acc, lane);
#pragma unroll
  for (int ct=0; ct<8; ++ct)
#pragma unroll
    for (int r=0;r<8;++r){
      int row = rowbase + r + 8*h, col = ct*16+rc;
      A2[(size_t)row*NHID+col] = acc[ct][r];
    }
}

// ---- fused edge kernel: 16 nodes/block, 144 edge rows, 9 waves --------------
// smem: sRel(144*42 f32) | sXq(672 f32) | sWsc(144 f32) | sNb(144 i32)
//     | sM(144*128 bf16) | sA(144*224 bf16) | wE1r(128*224) | wE2(128*128) | wX1(128*128)
#define EDGE_SMEM (((size_t)(144*42 + 16*42 + 144)*4) + 144*4 \
                   + ((size_t)144*128 + 144*224 + 128*224 + 2*128*128)*2)

__global__ void __launch_bounds__(288)
edge_kernel(const float* __restrict__ X, const int* __restrict__ idx,
            const float* __restrict__ A1p, const float* __restrict__ A2,
            const bf16* __restrict__ WT_e1r, const bf16* __restrict__ WT_e2,
            const bf16* __restrict__ WT_x1,
            const float* __restrict__ be2, const float* __restrict__ bx1,
            const float* __restrict__ Wx2, const float* __restrict__ bx2,
            float* __restrict__ Xout, bf16* __restrict__ MsumS){
  extern __shared__ char smem[];
  float* sRel = (float*)smem;            // [144][42]
  float* sXq  = sRel + 144*42;           // [16][42]
  float* sWsc = sXq + 16*42;             // [144]
  int*   sNb  = (int*)(sWsc + 144);      // [144]
  bf16*  sM   = (bf16*)(sNb + 144);      // [144][128]
  bf16*  sA   = sM + 144*128;            // [144][224]
  bf16*  wE1r = sA + 144*224;            // [128][224]
  bf16*  wE2  = wE1r + 128*224;          // [128][128]
  bf16*  wX1  = wE2 + 128*128;           // [128][128]

  const int b = blockIdx.y, n0 = blockIdx.x * 16;
  const int tid = threadIdx.x, wave = tid >> 5, lane = tid & 31;
  const int rc = lane & 15, h = lane >> 4;
  const int row0 = wave*16;

  // async-stage all three weight matrices; overlapped with rel/radial build
  stage_w_async(wE1r, WT_e1r, 128*224*2, tid, 288);
  stage_w_async(wE2,  WT_e2,  128*128*2, tid, 288);
  stage_w_async(wX1,  WT_x1,  128*128*2, tid, 288);

  for (int t_ = tid; t_ < 16*42; t_ += 288)
    sXq[t_] = X[(size_t)(b*NN + n0)*42 + t_];
  for (int t_ = tid; t_ < 144; t_ += 288)
    sNb[t_] = idx[(size_t)(b*NN + n0)*NKNN + t_];
  __syncthreads();

  // rel = X[b,node] - X[b,neighbor]
  if (lane < 16){
    int rr = row0 + lane, node = rr/NKNN, d = sNb[rr];
    const float* xd = X + (size_t)(b*NN + d)*42;
    const float* xq = sXq + node*42;
#pragma unroll 6
    for (int j=0;j<42;++j) sRel[rr*42 + j] = xq[j] - xd[j];
  }
  __syncthreads();

  // radial: silu(rel · relᵀ / C) -> bf16 A-tile (196 -> pad 224)
  {
    int rr = row0 + rc;
    const float* rel = sRel + rr*42;
    const int j0 = h*98;
    for (int jj=0;jj<98;++jj){
      int j = j0+jj, c_ = j/NCH, e_ = j%NCH;
      float s = rel[c_*3]*rel[e_*3] + rel[c_*3+1]*rel[e_*3+1] + rel[c_*3+2]*rel[e_*3+2];
      sA[(size_t)rr*224 + j] = (bf16)siluf(s * (1.0f/(float)NCH));
    }
    if (h==0)
      for (int j=196;j<224;++j) sA[(size_t)rr*224 + j] = (bf16)0.0f;
  }
  wait_async_all();
  __syncthreads();

  // h1 = silu( radial@We1_r + A1p[node] + A2[nbr] )
  v8f acc[8]; zeroacc<8>(acc);
  wmma_rowtile<7,8>(sA + (size_t)row0*224, 224, wE1r, 224, acc, lane);
#pragma unroll
  for (int r=0;r<8;++r){
    int m = r + 8*h, rr = row0 + m;
    int node = rr/NKNN, d = sNb[rr];
    const float* a1 = A1p + (size_t)(b*NN + n0 + node)*NHID;
    const float* a2 = A2  + (size_t)(b*NN + d)*NHID;
#pragma unroll
    for (int ct=0; ct<8; ++ct){
      int col = ct*16 + rc;
      sA[(size_t)rr*224 + col] = (bf16)siluf(acc[ct][r] + a1[col] + a2[col]);
    }
  }
  __syncthreads();

  // m = h1s @ We2 + be2 ; keep m (bf16) and silu(m)
  zeroacc<8>(acc);
  wmma_rowtile<4,8>(sA + (size_t)row0*224, 224, wE2, 128, acc, lane);
  __syncthreads();
#pragma unroll
  for (int ct=0; ct<8; ++ct)
#pragma unroll
    for (int r=0;r<8;++r){
      int m = r + 8*h, rr = row0 + m, col = ct*16+rc;
      float mv = acc[ct][r] + be2[col];
      sM[(size_t)rr*128 + col] = (bf16)mv;
      sA[(size_t)rr*224 + col] = (bf16)siluf(mv);
    }
  __syncthreads();

  // u = silu(m)@Wx1 + bx1 ; w = silu(u)·Wx2 + bx2
  zeroacc<8>(acc);
  wmma_rowtile<4,8>(sA + (size_t)row0*224, 224, wX1, 128, acc, lane);
  {
    float part[8];
#pragma unroll
    for (int r=0;r<8;++r) part[r]=0.0f;
#pragma unroll
    for (int ct=0; ct<8; ++ct){
      int col = ct*16+rc;
      float wx = Wx2[col], bb = bx1[col];
#pragma unroll
      for (int r=0;r<8;++r) part[r] += siluf(acc[ct][r] + bb) * wx;
    }
#pragma unroll
    for (int off=1; off<16; off<<=1)
#pragma unroll
      for (int r=0;r<8;++r) part[r] += __shfl_xor(part[r], off, 32);
    if (rc==0){
      float b2 = bx2[0];
#pragma unroll
      for (int r=0;r<8;++r) sWsc[row0 + r + 8*h] = part[r] + b2;
    }
  }
  __syncthreads();

  // node reductions (no atomics): silu(sum_k m) and X_out = X + mean_k rel*w
  for (int t_=tid; t_<16*128; t_+=288){
    int node = t_>>7, col = t_&127;
    float s = 0.0f;
#pragma unroll
    for (int k=0;k<NKNN;++k) s += (float)sM[(size_t)(node*NKNN+k)*128 + col];
    MsumS[(size_t)(b*NN + n0 + node)*128 + col] = (bf16)siluf(s);
  }
  for (int t_=tid; t_<16*42; t_+=288){
    int node = t_/42, j = t_%42;
    float s = 0.0f;
#pragma unroll
    for (int k=0;k<NKNN;++k) s += sRel[(node*NKNN+k)*42 + j] * sWsc[node*NKNN+k];
    Xout[(size_t)(b*NN + n0 + node)*42 + j] = sXq[node*42 + j] + s*(1.0f/(float)NKNN);
  }
}

// ---- Hn = Hh + ffn(concat[Hh, Msum]) -> silu(Hn) bf16 -----------------------
// smem: sbuf(4*16*128) | w1t | w1b | w2   = 114688 B
#define HN_SMEM ((size_t)(4*16*128 + 3*128*128)*2)
__global__ void hn_kernel(const bf16* __restrict__ sHh, const bf16* __restrict__ sMsum,
                          const bf16* __restrict__ WT_h1t, const bf16* __restrict__ WT_h1b,
                          const bf16* __restrict__ WT_h2,
                          const float* __restrict__ bh1, const float* __restrict__ bh2,
                          const float* __restrict__ Hh, bf16* __restrict__ sHn){
  extern __shared__ char smem[];
  bf16* sbuf = (bf16*)smem;
  bf16* w1t  = sbuf + 4*16*128;
  bf16* w1b  = w1t + 128*128;
  bf16* w2   = w1b + 128*128;
  const int tid = threadIdx.x;
  const int wave = tid >> 5, lane = tid & 31;
  const int rc = lane & 15, h = lane >> 4;
  const int rowbase = blockIdx.x*64 + wave*16;
  bf16* myb = sbuf + wave*16*128;

  stage_w_async(w1t, WT_h1t, 128*128*2, tid, 128);
  stage_w_async(w1b, WT_h1b, 128*128*2, tid, 128);
  stage_w_async(w2,  WT_h2,  128*128*2, tid, 128);
  wait_async_all();
  __syncthreads();

  v8f acc[8]; zeroacc<8>(acc);
  wmma_rowtile<4,8>(sHh   + (size_t)rowbase*NE, 128, w1t, 128, acc, lane);
  wmma_rowtile<4,8>(sMsum + (size_t)rowbase*NE, 128, w1b, 128, acc, lane);
#pragma unroll
  for (int ct=0; ct<8; ++ct)
#pragma unroll
    for (int r=0;r<8;++r){
      int col = ct*16+rc;
      myb[(r+8*h)*128 + col] = (bf16)siluf(acc[ct][r] + bh1[col]);
    }
  __syncthreads();
  zeroacc<8>(acc);
  wmma_rowtile<4,8>(myb, 128, w2, 128, acc, lane);
#pragma unroll
  for (int ct=0; ct<8; ++ct)
#pragma unroll
    for (int r=0;r<8;++r){
      int row = rowbase + r + 8*h, col = ct*16+rc;
      float v = acc[ct][r] + bh2[col] + Hh[(size_t)row*NE+col];
      sHn[(size_t)row*NE+col] = (bf16)siluf(v);
    }
}

// ---- gate->logits, P = siluHn@Wd1t+bd1, Q = siluHn@Wd1b ---------------------
__global__ void post_kernel(const bf16* __restrict__ sHn, const int* __restrict__ S,
                            const float* __restrict__ emb,
                            const bf16* __restrict__ WT_p1, const bf16* __restrict__ WT_r1,
                            const bf16* __restrict__ WT_r2, const bf16* __restrict__ WT_d1t,
                            const bf16* __restrict__ WT_d1b,
                            const float* __restrict__ bp1, const float* __restrict__ br1,
                            const float* __restrict__ br2, const float* __restrict__ bd1,
                            float* __restrict__ logits, bf16* __restrict__ Pb, bf16* __restrict__ Qb){
  __shared__ __attribute__((aligned(32))) bf16 sbuf[4][16][128];
  const int wave = threadIdx.x >> 5, lane = threadIdx.x & 31;
  const int rc = lane & 15, h = lane >> 4;
  const int rowbase = blockIdx.x*64 + wave*16;
  const bf16* A = sHn + (size_t)rowbase*NE;
  v8f acc[8];

  // gate = sigmoid(siluHn@Wp1 + bp1); z = silu(emb[S]*gate)
  zeroacc<8>(acc);
  wmma_rowtile<4,8>(A, 128, WT_p1, 128, acc, lane);
#pragma unroll
  for (int ct=0; ct<8; ++ct)
#pragma unroll
    for (int r=0;r<8;++r){
      int row = rowbase + r + 8*h, col = ct*16+rc;
      float gate = sigmf(acc[ct][r] + bp1[col]);
      float h0 = emb[(size_t)S[row]*NE + col];
      sbuf[wave][r+8*h][col] = (bf16)siluf(h0 * gate);
    }
  __syncthreads();
  zeroacc<8>(acc);
  wmma_rowtile<4,8>(&sbuf[wave][0][0], 128, WT_r1, 128, acc, lane);
  __syncthreads();
#pragma unroll
  for (int ct=0; ct<8; ++ct)
#pragma unroll
    for (int r=0;r<8;++r){
      int col = ct*16+rc;
      sbuf[wave][r+8*h][col] = (bf16)siluf(acc[ct][r] + br1[col]);
    }
  __syncthreads();
  {
    v8f acc2[2]; zeroacc<2>(acc2);
    wmma_rowtile<4,2>(&sbuf[wave][0][0], 128, WT_r2, 128, acc2, lane);
#pragma unroll
    for (int ct=0; ct<2; ++ct)
#pragma unroll
      for (int r=0;r<8;++r){
        int row = rowbase + r + 8*h, col = ct*16+rc;
        if (col < NCLS) logits[(size_t)row*NCLS + col] = acc2[ct][r] + br2[col];
      }
  }
  // P and Q (bf16)
  zeroacc<8>(acc);
  wmma_rowtile<4,8>(A, 128, WT_d1t, 128, acc, lane);
#pragma unroll
  for (int ct=0; ct<8; ++ct)
#pragma unroll
    for (int r=0;r<8;++r){
      int row = rowbase + r + 8*h, col = ct*16+rc;
      Pb[(size_t)row*NHID+col] = (bf16)(acc[ct][r] + bd1[col]);
    }
  zeroacc<8>(acc);
  wmma_rowtile<4,8>(A, 128, WT_d1b, 128, acc, lane);
#pragma unroll
  for (int ct=0; ct<8; ++ct)
#pragma unroll
    for (int r=0;r<8;++r){
      int row = rowbase + r + 8*h, col = ct*16+rc;
      Qb[(size_t)row*NHID+col] = (bf16)acc[ct][r];
    }
}

// ---- pd per edge: silu(P[n]+Q[d])·Wd2 + silu(P[d]+Q[n])·Wd2 + 2*bd2 ---------
__global__ void pd_kernel(const int* __restrict__ idx, const bf16* __restrict__ Pb,
                          const bf16* __restrict__ Qb, const float* __restrict__ Wd2,
                          const float* __restrict__ bd2, float* __restrict__ pd){
  const size_t e = (size_t)blockIdx.x*256 + threadIdx.x;
  if (e >= (size_t)BNR*NKNN) return;
  const size_t bn = e / NKNN;
  const size_t b  = bn >> 10;
  const size_t dn = b*NN + (size_t)idx[e];
  const bf16* Pn = Pb + bn*NHID; const bf16* Qn = Qb + bn*NHID;
  const bf16* Pd = Pb + dn*NHID; const bf16* Qd = Qb + dn*NHID;
  float s = 0.0f;
#pragma unroll 4
  for (int i=0;i<NHID;++i){
    float w2 = Wd2[i];
    s += siluf((float)Pn[i] + (float)Qd[i]) * w2;
    s += siluf((float)Pd[i] + (float)Qn[i]) * w2;
  }
  pd[e] = s + 2.0f*bd2[0];
}

// ============================================================================
extern "C" void kernel_launch(void* const* d_in, const int* in_sizes, int n_in,
                              void* d_out, int out_size, void* d_ws, size_t ws_size,
                              hipStream_t stream){
  (void)in_sizes; (void)n_in; (void)out_size; (void)ws_size;
  const float* X  =(const float*)d_in[0];  const int*   S  =(const int*)d_in[1];
  const float* t  =(const float*)d_in[2];  const float* emb=(const float*)d_in[3];
  const float* Wt1=(const float*)d_in[4];  const float* bt1=(const float*)d_in[5];
  const float* Wt2=(const float*)d_in[6];  const float* bt2=(const float*)d_in[7];
  const float* We1=(const float*)d_in[8];  const float* be1=(const float*)d_in[9];
  const float* We2=(const float*)d_in[10]; const float* be2=(const float*)d_in[11];
  const float* Wx1=(const float*)d_in[12]; const float* bx1=(const float*)d_in[13];
  const float* Wx2=(const float*)d_in[14]; const float* bx2=(const float*)d_in[15];
  const float* Wh1=(const float*)d_in[16]; const float* bh1=(const float*)d_in[17];
  const float* Wh2=(const float*)d_in[18]; const float* bh2=(const float*)d_in[19];
  const float* Wd1=(const float*)d_in[20]; const float* bd1=(const float*)d_in[21];
  const float* Wd2=(const float*)d_in[22]; const float* bd2=(const float*)d_in[23];
  const float* Wp1=(const float*)d_in[24]; const float* bp1=(const float*)d_in[25];
  const float* Wr1=(const float*)d_in[26]; const float* br1=(const float*)d_in[27];
  const float* Wr2=(const float*)d_in[28]; const float* br2=(const float*)d_in[29];

  char* ws = (char*)d_ws; size_t off = 0;
  auto carve = [&](size_t bytes)->char*{
    char* p = ws + off; off = (off + bytes + 255) & ~(size_t)255; return p;
  };
  bf16* WT_t1   = (bf16*)carve(128*128*2);
  bf16* WT_t2   = (bf16*)carve(128*128*2);
  bf16* WT_e1hi = (bf16*)carve(128*128*2);
  bf16* WT_e1hnb= (bf16*)carve(128*128*2);
  bf16* WT_e1r  = (bf16*)carve(128*224*2);
  bf16* WT_e2   = (bf16*)carve(128*128*2);
  bf16* WT_x1   = (bf16*)carve(128*128*2);
  bf16* WT_h1t  = (bf16*)carve(128*128*2);
  bf16* WT_h1b  = (bf16*)carve(128*128*2);
  bf16* WT_h2   = (bf16*)carve(128*128*2);
  bf16* WT_d1t  = (bf16*)carve(128*128*2);
  bf16* WT_d1b  = (bf16*)carve(128*128*2);
  bf16* WT_p1   = (bf16*)carve(128*128*2);
  bf16* WT_r1   = (bf16*)carve(128*128*2);
  bf16* WT_r2   = (bf16*)carve(32*128*2);

  int*   idxb  = (int*)  carve((size_t)BNR*NKNN*4);
  float* Hh    = (float*)carve((size_t)BNR*NE*4);
  bf16*  sHh   = (bf16*) carve((size_t)BNR*NE*2);
  float* A1p   = (float*)carve((size_t)BNR*NHID*4);
  float* A2b   = (float*)carve((size_t)BNR*NHID*4);
  bf16*  MsumS = (bf16*) carve((size_t)BNR*NHID*2);
  bf16*  sHn   = (bf16*) carve((size_t)BNR*NE*2);
  bf16*  Pb    = (bf16*) carve((size_t)BNR*NHID*2);
  bf16*  Qb    = (bf16*) carve((size_t)BNR*NHID*2);

  float* out_logits = (float*)d_out;
  float* out_X  = out_logits + (size_t)BNR*NCLS;
  float* out_pd = out_X + (size_t)BNR*NCH*3;

  auto pack = [&](const float* W, int ldW, int k0, int krows, int n0, int nvalid,
                  bf16* o, int nout, int kpad){
    int total = nout*kpad;
    pack_kernel<<<(total+255)/256, 256, 0, stream>>>(W, ldW, k0, krows, n0, nvalid, o, nout, kpad);
  };
  pack(Wt1,128,  0,128, 0,128, WT_t1,  128,128);
  pack(Wt2,128,  0,128, 0,128, WT_t2,  128,128);
  pack(We1,128,  0,128, 0,128, WT_e1hi,128,128);
  pack(We1,128,128,128, 0,128, WT_e1hnb,128,128);
  pack(We1,128,256,196, 0,128, WT_e1r, 128,224);
  pack(We2,128,  0,128, 0,128, WT_e2,  128,128);
  pack(Wx1,128,  0,128, 0,128, WT_x1,  128,128);
  pack(Wh1,128,  0,128, 0,128, WT_h1t, 128,128);
  pack(Wh1,128,128,128, 0,128, WT_h1b, 128,128);
  pack(Wh2,128,  0,128, 0,128, WT_h2,  128,128);
  pack(Wd1,128,  0,128, 0,128, WT_d1t, 128,128);
  pack(Wd1,128,128,128, 0,128, WT_d1b, 128,128);
  pack(Wp1,128,  0,128, 0,128, WT_p1,  128,128);
  pack(Wr1,128,  0,128, 0,128, WT_r1,  128,128);
  pack(Wr2, 25,  0,128, 0, 25, WT_r2,   32,128);

  knn_kernel<<<dim3(NN/256, NB), 256, 0, stream>>>(X, idxb);
  hh_kernel<<<BNR/64, 128, HH_SMEM, stream>>>(t, S, emb, bt1, bt2, WT_t1, WT_t2, Hh, sHh);
  a1a2_kernel<<<BNR/64, 128, A1A2_SMEM, stream>>>(sHh, WT_e1hi, WT_e1hnb, be1, A1p, A2b);
  edge_kernel<<<dim3(NN/16, NB), 288, EDGE_SMEM, stream>>>(
      X, idxb, A1p, A2b, WT_e1r, WT_e2, WT_x1, be2, bx1, Wx2, bx2, out_X, MsumS);
  hn_kernel<<<BNR/64, 128, HN_SMEM, stream>>>(sHh, MsumS, WT_h1t, WT_h1b, WT_h2, bh1, bh2, Hh, sHn);
  post_kernel<<<BNR/64, 128, 0, stream>>>(sHn, S, emb, WT_p1, WT_r1, WT_r2, WT_d1t, WT_d1b,
                                          bp1, br1, br2, bd1, out_logits, Pb, Qb);
  pd_kernel<<<((size_t)BNR*NKNN + 255)/256, 256, 0, stream>>>(idxb, Pb, Qb, Wd2, bd2, out_pd);
}
